// AttnBlockpp_70635032150266
// MI455X (gfx1250) — compile-verified, matrix operand order checked by reading, not run on hardware
//
#include <hip/hip_runtime.h>
#include <hip/hip_bf16.h>

// ---------------------------------------------------------------------------
// Types for WMMA fragments (wave32, gfx1250)
// ---------------------------------------------------------------------------
typedef __attribute__((ext_vector_type(16))) _Float16 v16h;
typedef __attribute__((ext_vector_type(8)))  _Float16 v8h;
typedef __attribute__((ext_vector_type(8)))  float    v8f;

#define WMMA_F16(a, b, c) \
    __builtin_amdgcn_wmma_f32_16x16x32_f16(false, (a), false, (b), (short)0, (c), false, false)

static __device__ __forceinline__ v8f v8f_zero() {
    v8f z = {0.f, 0.f, 0.f, 0.f, 0.f, 0.f, 0.f, 0.f};
    return z;
}

// Truncate a generic pointer to the 32-bit LDS byte address (ISA 10.2: the low
// 32 bits of a generic LDS address are the LDS offset).
static __device__ __forceinline__ unsigned lds_addr32(const void* p) {
    return (unsigned)(unsigned long long)p;
}

// CDNA5 async DMA: global -> LDS, 16 bytes per lane, tracked by ASYNCcnt.
static __device__ __forceinline__ void async_load_b128(unsigned lds_addr, const void* gaddr) {
    asm volatile("global_load_async_to_lds_b128 %0, %1, off"
                 :: "v"(lds_addr), "v"(gaddr) : "memory");
}
static __device__ __forceinline__ void wait_async0() {
    asm volatile("s_wait_asynccnt 0" ::: "memory");
}

// A-fragment (16x32 MxK f16) or B-fragment (32x16 KxN with column n contiguous
// in memory) loader.  ISA 7.12.2 layout:
//   lane&15 -> row (A) / column (B)
//   lane<16 -> K halves {0..7,16..23}; lane>=16 -> {8..15,24..31}
static __device__ __forceinline__ v16h load_frag_rm(const _Float16* base, int stride, int lane) {
    int r    = lane & 15;
    int koff = (lane & 16) ? 8 : 0;
    const _Float16* p = base + r * stride + koff;
    v8h lo = *(const v8h*)(p);
    v8h hi = *(const v8h*)(p + 16);
    v16h out;
#pragma unroll
    for (int i = 0; i < 8; ++i) { out[i] = lo[i]; out[i + 8] = hi[i]; }
    return out;
}

// ---------------------------------------------------------------------------
// Problem constants
// ---------------------------------------------------------------------------
#define BB   4
#define CC   256
#define NN   4096        // H*W
#define GG   32          // groups
#define CPG  8           // channels per group
#define KT_  32          // attention key-tile size
#define NKT  (NN / KT_)  // 128 key tiles

// log2(e) folded into q so softmax exponentials are native v_exp_f32 (exp2).
#define QSCALE (0.0625f * 1.44269504088896340736f)

// ---------------------------------------------------------------------------
// 1) GroupNorm statistics: one block per (b, g)
// ---------------------------------------------------------------------------
__global__ __launch_bounds__(256) void gn_stats_kernel(const float* __restrict__ x,
                                                       float* __restrict__ mean,
                                                       float* __restrict__ rstd) {
    int bg = blockIdx.x;
    const float* p = x + (size_t)bg * CPG * NN;
    float s = 0.f, ss = 0.f;
    for (int i = threadIdx.x; i < CPG * NN; i += 256) {
        float v = p[i];
        s += v; ss += v * v;
    }
    __shared__ float sh0[256], sh1[256];
    sh0[threadIdx.x] = s; sh1[threadIdx.x] = ss;
    __syncthreads();
    for (int o = 128; o > 0; o >>= 1) {
        if (threadIdx.x < o) {
            sh0[threadIdx.x] += sh0[threadIdx.x + o];
            sh1[threadIdx.x] += sh1[threadIdx.x + o];
        }
        __syncthreads();
    }
    if (threadIdx.x == 0) {
        float m   = sh0[0] * (1.f / (CPG * NN));
        float var = sh1[0] * (1.f / (CPG * NN)) - m * m;
        mean[bg] = m;
        rstd[bg] = rsqrtf(var + 1e-6f);
    }
}

// ---------------------------------------------------------------------------
// 2) Apply GN + transpose [B,C,N] -> hs f16 [B,N,C] via 32x32 LDS tile
// ---------------------------------------------------------------------------
__global__ __launch_bounds__(256) void gn_apply_kernel(const float* __restrict__ x,
                                                       const float* __restrict__ gw,
                                                       const float* __restrict__ gb,
                                                       const float* __restrict__ mean,
                                                       const float* __restrict__ rstd,
                                                       _Float16* __restrict__ hs) {
    int b  = blockIdx.z;
    int c0 = blockIdx.y * 32;
    int n0 = blockIdx.x * 32;
    __shared__ float tile[32][33];
    int tx = threadIdx.x, ty = threadIdx.y;
#pragma unroll
    for (int r = ty; r < 32; r += 8) {
        int c  = c0 + r;
        int bg = b * GG + (c >> 3);
        float sc = gw[c] * rstd[bg];
        float sf = gb[c] - mean[bg] * sc;
        tile[r][tx] = x[((size_t)(b * CC + c)) * NN + n0 + tx] * sc + sf;
    }
    __syncthreads();
#pragma unroll
    for (int r = ty; r < 32; r += 8) {
        int n = n0 + r;
        hs[((size_t)(b * NN + n)) * CC + c0 + tx] = (_Float16)tile[tx][r];
    }
}

// ---------------------------------------------------------------------------
// 3) f32 -> f16 weight conversion
// ---------------------------------------------------------------------------
__global__ __launch_bounds__(256) void cvt_kernel(const float* __restrict__ src,
                                                  _Float16* __restrict__ dst, int n) {
    int i = blockIdx.x * 256 + threadIdx.x;
    if (i < n) dst[i] = (_Float16)src[i];
}

// ---------------------------------------------------------------------------
// 4) QKV projection: Y = hs @ W^T + bias  (q scaled by C^-1/2 * log2e).
//    Block: 64 rows x 256 cols, 8 waves x (16 rows x 128 cols).
//    Double-buffered async A-tile staging; B frags batched before WMMA chain.
// ---------------------------------------------------------------------------
__global__ __launch_bounds__(256) void qkv_gemm_kernel(const _Float16* __restrict__ hs,
                                                       const _Float16* __restrict__ wh,
                                                       const float* __restrict__ bq,
                                                       const float* __restrict__ bk,
                                                       const float* __restrict__ bv,
                                                       _Float16* __restrict__ qo,
                                                       _Float16* __restrict__ ko,
                                                       _Float16* __restrict__ vo) {
    int mt   = blockIdx.x;
    int proj = blockIdx.y;
    int b    = blockIdx.z;
    const _Float16* W    = wh + (size_t)proj * CC * CC;
    const float*    bias = (proj == 0) ? bq : ((proj == 1) ? bk : bv);
    _Float16*       out  = (proj == 0) ? qo : ((proj == 1) ? ko : vo);
    float scale = (proj == 0) ? QSCALE : 1.0f;

    const _Float16* A = hs + ((size_t)b * NN + mt * 64) * CC;

    __shared__ _Float16 ldsA[2][64 * 40];   // 64 rows x 32 halves, pad 40
    int tid  = threadIdx.x;
    int lane = tid & 31;
    int wid  = tid >> 5;
    int m0w  = (wid & 3) * 16;
    int n0w  = (wid >> 2) * 128;

    int arow = tid >> 2, aoff = (tid & 3) * 8;   // one b128 chunk per thread
    const _Float16* Asrc = A + (size_t)arow * CC + aoff;

    v8f acc[8];
#pragma unroll
    for (int i = 0; i < 8; ++i) acc[i] = v8f_zero();

    // prologue: async-stage K-slab 0
    async_load_b128(lds_addr32(&ldsA[0][arow * 40 + aoff]), Asrc);
    wait_async0();
    __syncthreads();

    for (int kt = 0; kt < 8; ++kt) {
        int cur = kt & 1;
        if (kt < 7)   // prefetch next A slab into the other buffer
            async_load_b128(lds_addr32(&ldsA[cur ^ 1][arow * 40 + aoff]),
                            Asrc + (kt + 1) * 32);

        v16h af = load_frag_rm(&ldsA[cur][m0w * 40], 40, lane);
        v16h bf[8];
#pragma unroll
        for (int nt = 0; nt < 8; ++nt)
            bf[nt] = load_frag_rm(&W[(size_t)(n0w + nt * 16) * CC + kt * 32], CC, lane);
#pragma unroll
        for (int nt = 0; nt < 8; ++nt)
            acc[nt] = WMMA_F16(af, bf[nt], acc[nt]);

        if (kt < 7) { wait_async0(); __syncthreads(); }
    }

    int roff = (lane & 16) ? 8 : 0;
    int col  = lane & 15;
#pragma unroll
    for (int nt = 0; nt < 8; ++nt) {
        int n = n0w + nt * 16 + col;
        float bb = bias[n];
#pragma unroll
        for (int i = 0; i < 8; ++i) {
            int m = mt * 64 + m0w + roff + i;
            out[((size_t)b * NN + m) * CC + n] = (_Float16)((acc[nt][i] + bb) * scale);
        }
    }
}

// ---------------------------------------------------------------------------
// 5) Flash attention with online softmax (base-2, scale folded into q).
//    Block: 128 query rows (16/wave); 32-key K/V tiles, double-buffered:
//      K tile  -> LDS row-major via async DMA (ASYNCcnt)
//      V tile  -> LDS transposed (so P*V B-frags are contiguous b128 loads)
// ---------------------------------------------------------------------------
static __device__ __forceinline__ void stage_kv_tile(const _Float16* __restrict__ Ks,
                                                     const _Float16* __restrict__ Vs,
                                                     _Float16* ldsKbuf,
                                                     _Float16* ldsVTbuf, int tid) {
#pragma unroll
    for (int c = tid; c < 1024; c += 256) {          // 32 keys x 256 C in b128 chunks
        int row = c >> 5, off = (c & 31) * 8;        // row = key, off = channel
        async_load_b128(lds_addr32(ldsKbuf + row * 264 + off),
                        Ks + (size_t)row * CC + off);
        v8h vv = *(const v8h*)(Vs + (size_t)row * CC + off);
#pragma unroll
        for (int j2 = 0; j2 < 8; ++j2)
            ldsVTbuf[(off + j2) * 40 + row] = vv[j2];  // transpose: [C][key]
    }
}

__global__ __launch_bounds__(256) void attn_kernel(const _Float16* __restrict__ q,
                                                   const _Float16* __restrict__ k,
                                                   const _Float16* __restrict__ v,
                                                   _Float16* __restrict__ o) {
    int b    = blockIdx.y;
    int mt   = blockIdx.x;
    int tid  = threadIdx.x;
    int lane = tid & 31;
    int wid  = tid >> 5;

    __shared__ _Float16 ldsK[2][32 * 264];    // 32 keys x 256, pad 264
    __shared__ _Float16 ldsVT[2][256 * 40];   // 256 C x 32 keys, pad 40
    __shared__ _Float16 ldsP[8][16 * 40];     // per-wave 16x32 P, pad 40

    const _Float16* Kb0 = k + (size_t)b * NN * CC;
    const _Float16* Vb0 = v + (size_t)b * NN * CC;

    // Q fragments: register resident for the whole pass.
    const _Float16* Qb = q + ((size_t)b * NN + mt * 128 + wid * 16) * CC;
    v16h qf[8];
#pragma unroll
    for (int kt = 0; kt < 8; ++kt) qf[kt] = load_frag_rm(Qb + kt * 32, CC, lane);

    v8f acco[16];
#pragma unroll
    for (int i = 0; i < 16; ++i) acco[i] = v8f_zero();
    float mrow[8], lrow[8];
#pragma unroll
    for (int i = 0; i < 8; ++i) { mrow[i] = -1e30f; lrow[i] = 0.f; }

    int roff = (lane & 16) ? 8 : 0;
    int col  = lane & 15;

    // prologue: stage tile 0
    stage_kv_tile(Kb0, Vb0, ldsK[0], ldsVT[0], tid);
    wait_async0();
    __syncthreads();

    for (int j = 0; j < NKT; ++j) {
        int cur = j & 1;
        if (j + 1 < NKT)   // prefetch next tile into the other buffers
            stage_kv_tile(Kb0 + (size_t)(j + 1) * KT_ * CC,
                          Vb0 + (size_t)(j + 1) * KT_ * CC,
                          ldsK[cur ^ 1], ldsVT[cur ^ 1], tid);

        // ---- S = q k^T (16 x 32), fragments batched ahead of WMMA chains
        v8f s0 = v8f_zero(), s1 = v8f_zero();
#pragma unroll
        for (int nt = 0; nt < 2; ++nt) {
            v16h bf[8];
#pragma unroll
            for (int kt = 0; kt < 8; ++kt)
                bf[kt] = load_frag_rm(&ldsK[cur][nt * 16 * 264 + kt * 32], 264, lane);
            if (nt == 0) {
#pragma unroll
                for (int kt = 0; kt < 8; ++kt) s0 = WMMA_F16(qf[kt], bf[kt], s0);
            } else {
#pragma unroll
                for (int kt = 0; kt < 8; ++kt) s1 = WMMA_F16(qf[kt], bf[kt], s1);
            }
        }

        // ---- online softmax (base-2; q pre-scaled by log2e/sqrt(C))
        float tm[8];
#pragma unroll
        for (int i = 0; i < 8; ++i) tm[i] = fmaxf(s0[i], s1[i]);
#pragma unroll
        for (int msk = 1; msk < 16; msk <<= 1)
#pragma unroll
            for (int i = 0; i < 8; ++i) tm[i] = fmaxf(tm[i], __shfl_xor(tm[i], msk, 32));

        float corr[8];
#pragma unroll
        for (int i = 0; i < 8; ++i) {
            float mn = fmaxf(mrow[i], tm[i]);
            corr[i]  = __builtin_amdgcn_exp2f(mrow[i] - mn);
            mrow[i]  = mn;
        }
        float ts[8];
#pragma unroll
        for (int i = 0; i < 8; ++i) {
            float p0 = __builtin_amdgcn_exp2f(s0[i] - mrow[i]);
            float p1 = __builtin_amdgcn_exp2f(s1[i] - mrow[i]);
            s0[i] = p0; s1[i] = p1;
            ts[i] = p0 + p1;
        }
#pragma unroll
        for (int msk = 1; msk < 16; msk <<= 1)
#pragma unroll
            for (int i = 0; i < 8; ++i) ts[i] += __shfl_xor(ts[i], msk, 32);
#pragma unroll
        for (int i = 0; i < 8; ++i) lrow[i] = lrow[i] * corr[i] + ts[i];
#pragma unroll
        for (int nt = 0; nt < 16; ++nt)
#pragma unroll
            for (int i = 0; i < 8; ++i) acco[nt][i] *= corr[i];

        // ---- P -> per-wave LDS -> A fragment (same-wave DS ops stay in order)
        _Float16* Pw = ldsP[wid];
#pragma unroll
        for (int i = 0; i < 8; ++i) {
            Pw[(roff + i) * 40 + 0 * 16 + col] = (_Float16)s0[i];
            Pw[(roff + i) * 40 + 1 * 16 + col] = (_Float16)s1[i];
        }
        v16h pa = load_frag_rm(Pw, 40, lane);

        // ---- O += P * V; V^T rows are contiguous, 16 independent chains
#pragma unroll
        for (int g = 0; g < 4; ++g) {
            v16h vb[4];
#pragma unroll
            for (int t = 0; t < 4; ++t)
                vb[t] = load_frag_rm(&ldsVT[cur][(g * 4 + t) * 16 * 40], 40, lane);
#pragma unroll
            for (int t = 0; t < 4; ++t)
                acco[g * 4 + t] = WMMA_F16(pa, vb[t], acco[g * 4 + t]);
        }

        if (j + 1 < NKT) { wait_async0(); __syncthreads(); }
    }

    float linv[8];
#pragma unroll
    for (int i = 0; i < 8; ++i) linv[i] = 1.f / lrow[i];
#pragma unroll
    for (int nt = 0; nt < 16; ++nt)
#pragma unroll
        for (int i = 0; i < 8; ++i) {
            int m = mt * 128 + wid * 16 + roff + i;
            o[((size_t)b * NN + m) * CC + nt * 16 + col] = (_Float16)(acco[nt][i] * linv[i]);
        }
}

// ---------------------------------------------------------------------------
// 6) Output projection + residual: out = (x + ao @ Wo^T + bo) * 2^-1/2
// ---------------------------------------------------------------------------
__global__ __launch_bounds__(256) void out_proj_kernel(const _Float16* __restrict__ ao,
                                                       const _Float16* __restrict__ woh,
                                                       const float* __restrict__ bo,
                                                       const float* __restrict__ x,
                                                       float* __restrict__ out) {
    int mt = blockIdx.x;
    int b  = blockIdx.z;
    const _Float16* A = ao + ((size_t)b * NN + mt * 64) * CC;

    __shared__ _Float16 ldsA[2][64 * 40];
    int tid  = threadIdx.x;
    int lane = tid & 31;
    int wid  = tid >> 5;
    int m0w  = (wid & 3) * 16;
    int n0w  = (wid >> 2) * 128;

    int arow = tid >> 2, aoff = (tid & 3) * 8;
    const _Float16* Asrc = A + (size_t)arow * CC + aoff;

    v8f acc[8];
#pragma unroll
    for (int i = 0; i < 8; ++i) acc[i] = v8f_zero();

    async_load_b128(lds_addr32(&ldsA[0][arow * 40 + aoff]), Asrc);
    wait_async0();
    __syncthreads();

    for (int kt = 0; kt < 8; ++kt) {
        int cur = kt & 1;
        if (kt < 7)
            async_load_b128(lds_addr32(&ldsA[cur ^ 1][arow * 40 + aoff]),
                            Asrc + (kt + 1) * 32);

        v16h af = load_frag_rm(&ldsA[cur][m0w * 40], 40, lane);
        v16h bf[8];
#pragma unroll
        for (int nt = 0; nt < 8; ++nt)
            bf[nt] = load_frag_rm(&woh[(size_t)(n0w + nt * 16) * CC + kt * 32], CC, lane);
#pragma unroll
        for (int nt = 0; nt < 8; ++nt)
            acc[nt] = WMMA_F16(af, bf[nt], acc[nt]);

        if (kt < 7) { wait_async0(); __syncthreads(); }
    }

    int roff = (lane & 16) ? 8 : 0;
    int col  = lane & 15;
    const float inv_sqrt2 = 0.70710678118654752f;
#pragma unroll
    for (int nt = 0; nt < 8; ++nt) {
        int n = n0w + nt * 16 + col;           // output channel
        float bb = bo[n];
#pragma unroll
        for (int i = 0; i < 8; ++i) {
            int m = mt * 64 + m0w + roff + i;  // pixel index
            size_t xi = ((size_t)(b * CC + n)) * NN + m;
            out[xi] = (x[xi] + acc[nt][i] + bb) * inv_sqrt2;
        }
    }
}

// ---------------------------------------------------------------------------
// Host launcher
// ---------------------------------------------------------------------------
extern "C" void kernel_launch(void* const* d_in, const int* in_sizes, int n_in,
                              void* d_out, int out_size, void* d_ws, size_t ws_size,
                              hipStream_t stream) {
    const float* x  = (const float*)d_in[0];
    const float* gw = (const float*)d_in[1];
    const float* gb = (const float*)d_in[2];
    const float* Wq = (const float*)d_in[3];
    const float* bq = (const float*)d_in[4];
    const float* Wk = (const float*)d_in[5];
    const float* bk = (const float*)d_in[6];
    const float* Wv = (const float*)d_in[7];
    const float* bv = (const float*)d_in[8];
    const float* Wo = (const float*)d_in[9];
    const float* bo = (const float*)d_in[10];
    float* out = (float*)d_out;

    // Workspace layout (16B-aligned); ~42 MB total, fits easily in HBM/L2.
    char* ws = (char*)d_ws;
    float*    mean = (float*)ws;
    float*    rstd = (float*)(ws + 512);
    _Float16* hs   = (_Float16*)(ws + 1024);
    _Float16* wh   = (_Float16*)(ws + 1024 + 8388608);
    _Float16* qb   = (_Float16*)(ws + 1024 + 8388608 + 524288);
    _Float16* kb   = qb + (size_t)BB * NN * CC;
    _Float16* vb   = kb + (size_t)BB * NN * CC;
    _Float16* ao   = vb + (size_t)BB * NN * CC;

    cvt_kernel<<<256, 256, 0, stream>>>(Wq, wh + 0 * CC * CC, CC * CC);
    cvt_kernel<<<256, 256, 0, stream>>>(Wk, wh + 1 * CC * CC, CC * CC);
    cvt_kernel<<<256, 256, 0, stream>>>(Wv, wh + 2 * CC * CC, CC * CC);
    cvt_kernel<<<256, 256, 0, stream>>>(Wo, wh + 3 * CC * CC, CC * CC);

    gn_stats_kernel<<<BB * GG, 256, 0, stream>>>(x, mean, rstd);
    gn_apply_kernel<<<dim3(NN / 32, CC / 32, BB), dim3(32, 8), 0, stream>>>(
        x, gw, gb, mean, rstd, hs);

    qkv_gemm_kernel<<<dim3(NN / 64, 3, BB), 256, 0, stream>>>(
        hs, wh, bq, bk, bv, qb, kb, vb);

    attn_kernel<<<dim3(NN / 128, BB), 256, 0, stream>>>(qb, kb, vb, ao);

    out_proj_kernel<<<dim3(NN / 64, 1, BB), 256, 0, stream>>>(
        ao, wh + 3 * CC * CC, bo, x, out);
}
